// GTAM_2D_22196390986138
// MI455X (gfx1250) — matrix operand order repaired and checked by reading, not compiled, not run on hardware
//
#include <hip/hip_runtime.h>
#include <hip/hip_bf16.h>

// ---------------------------------------------------------------------------
// GIN (OGB mol) forward on MI455X / gfx1250.
//   N=50000 nodes, E=640000 edges, D=128, 5 layers.
//   - message passing: fused gather + relu + global_atomic_add_f32 scatter
//     (h table = 25.6MB -> L2 resident on the 192MB L2; no [E,128] msg buffer)
//   - GEMMs: bf16 WMMA (v_wmma_f32_16x16x32_bf16), weights transposed in LDS,
//     B-fragments register-resident per block, A built/fused on the fly.
//   - BatchNorm stats fused into GEMM epilogues (LDS atomics -> 1 global
//     atomic per column per block), tiny finalize kernel folds gamma/beta.
// ---------------------------------------------------------------------------

typedef __attribute__((ext_vector_type(16))) __bf16 v16bf;
typedef __attribute__((ext_vector_type(8)))  __bf16 v8bf;
typedef __attribute__((ext_vector_type(8)))  float  v8f;

#define N_NODES 50000
#define N_EDGES 640000
#define D    128
#define D2   256
#define M_TILES (N_NODES / 16)   // 3125 exactly
#define GEMM_BLOCKS 625

__device__ __forceinline__ v16bf cat8(v8bf lo, v8bf hi) {
  v16bf r;
#pragma unroll
  for (int i = 0; i < 8; ++i) { r[i] = lo[i]; r[i + 8] = hi[i]; }
  return r;
}

__device__ __forceinline__ v8f wmma_bf16(v16bf a, v16bf b, v8f c) {
  // D = A(16x32 bf16) * B(32x16 bf16) + C(16x16 f32)
  return __builtin_amdgcn_wmma_f32_16x16x32_bf16(
      /*neg_a=*/false, a, /*neg_b=*/false, b,
      /*c_mod=*/(short)0, c, /*reuse_a=*/false, /*reuse_b=*/false);
}

// ---------------------------------------------------------------------------
// AtomEncoder: h[n,:] = sum_f atom_emb[f, x[n,f], :]   (one wave per node)
// ---------------------------------------------------------------------------
__global__ __launch_bounds__(256) void atom_encoder_kernel(
    const int* __restrict__ x, const float* __restrict__ atom_emb,
    float* __restrict__ h) {
  const int t = blockIdx.x * 256 + threadIdx.x;   // N*32 threads
  const int n = t >> 5;
  const int d0 = (t & 31) << 2;
  if (n >= N_NODES) return;
  float4 acc = make_float4(0.f, 0.f, 0.f, 0.f);
#pragma unroll
  for (int f = 0; f < 9; ++f) {
    const int v = x[n * 9 + f];
    const float4 e = *(const float4*)&atom_emb[((f * 119 + v) << 7) + d0];
    acc.x += e.x; acc.y += e.y; acc.z += e.z; acc.w += e.w;
  }
  *(float4*)&h[(n << 7) + d0] = acc;
}

// ---------------------------------------------------------------------------
// Edge kernel: aggr[dst,:] += relu(h[src,:] + bond_emb(edge_attr))
// One wave per edge; 4 f32 atomics per lane. h gathers hit L2 (25.6MB table).
// ---------------------------------------------------------------------------
__global__ __launch_bounds__(256) void edge_message_kernel(
    const int* __restrict__ edge_index, const int* __restrict__ edge_attr,
    const float* __restrict__ bond_emb_l,    // [3,6,128] layer slice
    const float* __restrict__ h, float* __restrict__ aggr) {
  const int t = blockIdx.x * 256 + threadIdx.x;
  const int e = t >> 5;
  const int d0 = (t & 31) << 2;
  if (e >= N_EDGES) return;
  const int src = edge_index[e];
  const int dst = edge_index[N_EDGES + e];
  const int a0 = edge_attr[e * 3 + 0];
  const int a1 = edge_attr[e * 3 + 1];
  const int a2 = edge_attr[e * 3 + 2];
  const float4 e0 = *(const float4*)&bond_emb_l[((0 * 6 + a0) << 7) + d0];
  const float4 e1 = *(const float4*)&bond_emb_l[((1 * 6 + a1) << 7) + d0];
  const float4 e2 = *(const float4*)&bond_emb_l[((2 * 6 + a2) << 7) + d0];
  const float4 hs = *(const float4*)&h[(src << 7) + d0];
  float4 m;
  m.x = fmaxf(hs.x + e0.x + e1.x + e2.x, 0.f);
  m.y = fmaxf(hs.y + e0.y + e1.y + e2.y, 0.f);
  m.z = fmaxf(hs.z + e0.z + e1.z + e2.z, 0.f);
  m.w = fmaxf(hs.w + e0.w + e1.w + e2.w, 0.f);
  float* out = &aggr[(dst << 7) + d0];
  unsafeAtomicAdd(out + 0, m.x);   // global_atomic_add_f32
  unsafeAtomicAdd(out + 1, m.y);
  unsafeAtomicAdd(out + 2, m.z);
  unsafeAtomicAdd(out + 3, m.w);
}

// ---------------------------------------------------------------------------
// GEMM1: z1[N,256] = bf16( (1+eps)*h + aggr ) @ W1[128,256] + b1
//        + fused per-column sum / sum-of-squares for BN1.
// 256 threads = 8 waves; wave w owns columns [32w, 32w+32) (two 16-col tiles).
// W1 transposed into LDS once per block; B frags live in registers.
// ---------------------------------------------------------------------------
__global__ __launch_bounds__(256) void gemm1_kernel(
    const float* __restrict__ h, const float* __restrict__ aggr,
    const float* __restrict__ W1l, const float* __restrict__ b1l,
    const float* __restrict__ eps_l,
    float* __restrict__ z1,
    float* __restrict__ Ssum, float* __restrict__ Sssq) {
  constexpr int K = 128, NO = 256, KS = K + 8;   // KS: bank-rotating stride
  __shared__ __bf16 sWt[NO * KS];                // W^T, [n][k], 69.6KB
  __shared__ __bf16 sA[16 * KS];                 // one 16-row A tile
  __shared__ float  sS[NO];
  __shared__ float  sQ[NO];

  const int tid  = threadIdx.x;
  const int wave = tid >> 5;
  const int lane = tid & 31;
  const int nl   = lane & 15;
  const int kh   = lane >> 4;

  sS[tid] = 0.f;
  sQ[tid] = 0.f;

  // transpose-load W1 (f32 -> bf16) into LDS
  for (int idx = tid; idx < K * NO; idx += 256) {
    const int k = idx >> 8;          // / 256
    const int n = idx & (NO - 1);
    sWt[n * KS + k] = (__bf16)W1l[idx];
  }
  __syncthreads();

  // B fragments (contiguous k order, same as A: lanes 0-15 K=0..15 of col n; lanes 16-31 K=16..31)
  v16bf Bf[2][4];
#pragma unroll
  for (int t = 0; t < 2; ++t) {
    const int n0 = wave * 32 + t * 16 + nl;
#pragma unroll
    for (int kb = 0; kb < 4; ++kb) {
      const int ks = kb * 32 + kh * 16;
      const v8bf* p = (const v8bf*)&sWt[n0 * KS + ks];
      Bf[t][kb] = cat8(p[0], p[1]);
    }
  }

  const float ep    = 1.0f + eps_l[0];
  const int   colA  = wave * 32 + nl;
  const int   colB  = colA + 16;
  const float biasA = b1l[colA];
  const float biasB = b1l[colB];

  for (int mt = blockIdx.x; mt < M_TILES; mt += gridDim.x) {
    const int row0  = mt * 16;
    const int mtn   = mt + (int)gridDim.x;
    const int nrow0 = (mtn < M_TILES) ? mtn * 16 : row0;

    // A-build: x = (1+eps)*h + aggr  -> bf16 LDS tile; prefetch next tile
    for (int idx = tid; idx < 16 * K; idx += 256) {
      const int r = idx >> 7;
      const int c = idx & (K - 1);
      const int off = ((row0 + r) << 7) + c;
      sA[r * KS + c] = (__bf16)(ep * h[off] + aggr[off]);
      const int poff = ((nrow0 + r) << 7) + c;
      __builtin_prefetch(&h[poff], 0, 0);      // global_prefetch_b8
      __builtin_prefetch(&aggr[poff], 0, 0);
    }
    __syncthreads();

    v8f acc0 = {};
    v8f acc1 = {};
#pragma unroll
    for (int kb = 0; kb < 4; ++kb) {
      // A layout: lane m=nl; K runs {kh*16 .. +15} per kb (matches B)
      const int k1 = kb * 32 + kh * 16;
      const v8bf* pa = (const v8bf*)&sA[nl * KS + k1];
      const v16bf a = cat8(pa[0], pa[1]);      // contiguous 16 elements
      acc0 = wmma_bf16(a, Bf[0][kb], acc0);
      acc1 = wmma_bf16(a, Bf[1][kb], acc1);
    }

    // epilogue: +bias, store z1, BN stats (C layout: VGPR j -> M = kh*8+j)
    float s0 = 0.f, q0 = 0.f, s1 = 0.f, q1 = 0.f;
#pragma unroll
    for (int j = 0; j < 8; ++j) {
      const int row = row0 + kh * 8 + j;
      const float v0 = acc0[j] + biasA;
      const float v1 = acc1[j] + biasB;
      z1[row * NO + colA] = v0;
      z1[row * NO + colB] = v1;
      s0 += v0; q0 += v0 * v0;
      s1 += v1; q1 += v1 * v1;
    }
    atomicAdd(&sS[colA], s0);    // ds_add_f32
    atomicAdd(&sQ[colA], q0);
    atomicAdd(&sS[colB], s1);
    atomicAdd(&sQ[colB], q1);
    __syncthreads();
  }
  unsafeAtomicAdd(&Ssum[tid], sS[tid]);
  unsafeAtomicAdd(&Sssq[tid], sQ[tid]);
}

// ---------------------------------------------------------------------------
// GEMM2: z2[N,128] = bf16( relu(BN1(z1)) ) @ W2[256,128] + b2
//        + fused BN2 stats. Wave w owns 16 columns [16w, 16w+16).
// ---------------------------------------------------------------------------
__global__ __launch_bounds__(256) void gemm2_kernel(
    const float* __restrict__ z1,
    const float* __restrict__ scale1, const float* __restrict__ shift1,
    const float* __restrict__ W2l, const float* __restrict__ b2l,
    float* __restrict__ z2,
    float* __restrict__ Ssum, float* __restrict__ Sssq) {
  constexpr int K = 256, NO = 128, KS = K + 8;
  __shared__ __bf16 sWt[NO * KS];                // 67.6KB
  __shared__ __bf16 sA[16 * KS];
  __shared__ float  sS[NO];
  __shared__ float  sQ[NO];

  const int tid  = threadIdx.x;
  const int wave = tid >> 5;
  const int lane = tid & 31;
  const int nl   = lane & 15;
  const int kh   = lane >> 4;

  if (tid < NO) { sS[tid] = 0.f; sQ[tid] = 0.f; }

  for (int idx = tid; idx < K * NO; idx += 256) {
    const int k = idx >> 7;          // / 128
    const int n = idx & (NO - 1);
    sWt[n * KS + k] = (__bf16)W2l[idx];
  }
  __syncthreads();

  v16bf Bf[8];
  const int n0 = wave * 16 + nl;
#pragma unroll
  for (int kb = 0; kb < 8; ++kb) {
    const int ks = kb * 32 + kh * 16;
    const v8bf* p = (const v8bf*)&sWt[n0 * KS + ks];
    Bf[kb] = cat8(p[0], p[1]);
  }

  const int   col  = wave * 16 + nl;
  const float bias = b2l[col];

  for (int mt = blockIdx.x; mt < M_TILES; mt += gridDim.x) {
    const int row0 = mt * 16;

    // A-build: relu(z1 * scale1 + shift1) -> bf16 LDS tile
    for (int idx = tid; idx < 16 * K; idx += 256) {
      const int r = idx >> 8;
      const int c = idx & (K - 1);
      const float zv = z1[(row0 + r) * K + c];
      sA[r * KS + c] = (__bf16)fmaxf(zv * scale1[c] + shift1[c], 0.f);
    }
    __syncthreads();

    v8f acc = {};
#pragma unroll
    for (int kb = 0; kb < 8; ++kb) {
      const int k1 = kb * 32 + kh * 16;
      const v8bf* pa = (const v8bf*)&sA[nl * KS + k1];
      const v16bf a = cat8(pa[0], pa[1]);
      acc = wmma_bf16(a, Bf[kb], acc);
    }

    float s = 0.f, q = 0.f;
#pragma unroll
    for (int j = 0; j < 8; ++j) {
      const int row = row0 + kh * 8 + j;
      const float v = acc[j] + bias;
      z2[row * NO + col] = v;
      s += v; q += v * v;
    }
    atomicAdd(&sS[col], s);
    atomicAdd(&sQ[col], q);
    __syncthreads();
  }
  if (tid < NO) {
    unsafeAtomicAdd(&Ssum[tid], sS[tid]);
    unsafeAtomicAdd(&Sssq[tid], sQ[tid]);
  }
}

// ---------------------------------------------------------------------------
// Fold BN stats + gamma/beta into scale/shift: y = x*scale + shift
// ---------------------------------------------------------------------------
__global__ void bn_finalize_kernel(
    const float* __restrict__ S, const float* __restrict__ Q,
    const float* __restrict__ g, const float* __restrict__ b,
    float* __restrict__ scale, float* __restrict__ shift, int C) {
  const int c = threadIdx.x;
  if (c >= C) return;
  const float invN = 1.0f / (float)N_NODES;
  const float mean = S[c] * invN;
  const float var  = Q[c] * invN - mean * mean;   // biased variance (torch BN)
  const float sc   = g[c] * rsqrtf(var + 1e-5f);
  scale[c] = sc;
  shift[c] = b[c] - mean * sc;
}

// ---------------------------------------------------------------------------
// h = [relu]( z2 * scale2 + shift2 )
// ---------------------------------------------------------------------------
__global__ __launch_bounds__(256) void update_h_kernel(
    const float* __restrict__ z2, const float* __restrict__ scale,
    const float* __restrict__ shift, float* __restrict__ h, int do_relu) {
  const int t = blockIdx.x * 256 + threadIdx.x;
  const int n = t >> 5;
  const int d0 = (t & 31) << 2;
  if (n >= N_NODES) return;
  const float4 z = *(const float4*)&z2[(n << 7) + d0];
  float4 v;
  v.x = z.x * scale[d0 + 0] + shift[d0 + 0];
  v.y = z.y * scale[d0 + 1] + shift[d0 + 1];
  v.z = z.z * scale[d0 + 2] + shift[d0 + 2];
  v.w = z.w * scale[d0 + 3] + shift[d0 + 3];
  if (do_relu) {
    v.x = fmaxf(v.x, 0.f); v.y = fmaxf(v.y, 0.f);
    v.z = fmaxf(v.z, 0.f); v.w = fmaxf(v.w, 0.f);
  }
  *(float4*)&h[(n << 7) + d0] = v;
}

// ---------------------------------------------------------------------------
extern "C" void kernel_launch(void* const* d_in, const int* in_sizes, int n_in,
                              void* d_out, int out_size, void* d_ws, size_t ws_size,
                              hipStream_t stream) {
  (void)in_sizes; (void)n_in; (void)out_size; (void)ws_size;
  const int*   x        = (const int*)d_in[0];
  const int*   ei       = (const int*)d_in[1];
  const int*   ea       = (const int*)d_in[2];
  const float* atom_emb = (const float*)d_in[3];
  const float* bond_emb = (const float*)d_in[4];   // [5,3,6,128]
  const float* eps      = (const float*)d_in[5];
  const float* W1       = (const float*)d_in[6];   // [5,128,256]
  const float* b1       = (const float*)d_in[7];
  const float* g1       = (const float*)d_in[8];
  const float* be1      = (const float*)d_in[9];
  const float* W2       = (const float*)d_in[10];  // [5,256,128]
  const float* b2       = (const float*)d_in[11];
  const float* bn_g     = (const float*)d_in[12];
  const float* bn_b     = (const float*)d_in[13];
  float* h = (float*)d_out;                        // h lives in d_out

  // workspace layout (floats)
  float* ws   = (float*)d_ws;
  float* aggr = ws;                                  // N*128
  float* z1   = aggr + (size_t)N_NODES * D;          // N*256
  float* z2   = z1   + (size_t)N_NODES * D2;         // N*128
  float* S1   = z2   + (size_t)N_NODES * D;          // 256
  float* Q1   = S1 + D2;                             // 256
  float* S2   = Q1 + D2;                             // 128
  float* Q2   = S2 + D;                              // 128
  float* sc1  = Q2 + D;                              // 256
  float* sh1  = sc1 + D2;                            // 256
  float* sc2  = sh1 + D2;                            // 128
  float* sh2  = sc2 + D;                             // 128

  const int node_blocks = (N_NODES * 32) / 256;      // 6250
  const int edge_blocks = (N_EDGES * 32) / 256;      // 80000

  atom_encoder_kernel<<<node_blocks, 256, 0, stream>>>(x, atom_emb, h);

  for (int l = 0; l < 5; ++l) {
    hipMemsetAsync(aggr, 0, sizeof(float) * (size_t)N_NODES * D, stream);
    hipMemsetAsync(S1, 0, sizeof(float) * (2 * D2 + 2 * D), stream);

    edge_message_kernel<<<edge_blocks, 256, 0, stream>>>(
        ei, ea, bond_emb + (size_t)l * 3 * 6 * D, h, aggr);

    gemm1_kernel<<<GEMM_BLOCKS, 256, 0, stream>>>(
        h, aggr, W1 + (size_t)l * D * D2, b1 + (size_t)l * D2, eps + l,
        z1, S1, Q1);

    bn_finalize_kernel<<<1, D2, 0, stream>>>(
        S1, Q1, g1 + (size_t)l * D2, be1 + (size_t)l * D2, sc1, sh1, D2);

    gemm2_kernel<<<GEMM_BLOCKS, 256, 0, stream>>>(
        z1, sc1, sh1, W2 + (size_t)l * D2 * D, b2 + (size_t)l * D,
        z2, S2, Q2);

    bn_finalize_kernel<<<1, D, 0, stream>>>(
        S2, Q2, bn_g + (size_t)l * D, bn_b + (size_t)l * D, sc2, sh2, D);

    update_h_kernel<<<node_blocks, 256, 0, stream>>>(
        z2, sc2, sh2, h, (l < 4) ? 1 : 0);
  }
}
